// LSTM_66709432042072
// MI455X (gfx1250) — matrix operand order repaired and checked by reading, not compiled, not run on hardware
//
#include <hip/hip_runtime.h>
#include <hip/hip_bf16.h>
#include <math.h>

// ---------------------------------------------------------------------------
// LSTM forward for MI455X (gfx1250, wave32, WMMA bf16 16x16x32, f32 accum)
// B=256 batch, S=128 seq, D=256 embed, H=1024 hidden, C=256 classes
// Step kernel: weight slabs staged into a double-buffered LDS tile with
// gfx1250 async global->LDS DMA (ASYNCcnt), shared by 16 waves per block.
// ---------------------------------------------------------------------------

typedef __bf16 bf16_t;
typedef __attribute__((ext_vector_type(16))) __bf16 v16bf;
typedef __attribute__((ext_vector_type(8)))  __bf16 v8bf;
typedef __attribute__((ext_vector_type(8)))  float  v8f;

#define Bn 256
#define Sn 128
#define Dn 256
#define Hn 1024
#define Cn 256
#define AROW 40                 // padded LDS row stride -> conflict-free b128 reads
#define SLAB (4 * 16 * AROW)    // one K-step slab: 4 gates x 16 rows x 32 K (padded)

// D = A(16x32 bf16) * B(32x16 bf16) + C(16x16 f32)
static __device__ __forceinline__ v8f wmma_bf16(v16bf a, v16bf b, v8f c) {
  return __builtin_amdgcn_wmma_f32_16x16x32_bf16(
      /*neg_a=*/false, a, /*neg_b=*/false, b,
      /*c_mod=*/(short)0, c, /*reuse_a=*/false, /*reuse_b=*/false);
}

// A operand from a 32-wide row (LDS or global): per-lane row, hi = lane>>4.
static __device__ __forceinline__ v16bf load_a16(const bf16_t* rowPtr, int hi) {
  v8bf c0 = *(const v8bf*)(rowPtr + 8 * hi);
  v8bf c1 = *(const v8bf*)(rowPtr + 16 + 8 * hi);
  return __builtin_shufflevector(c0, c1, 0, 1, 2, 3, 4, 5, 6, 7,
                                 8, 9, 10, 11, 12, 13, 14, 15);
}

// B operand: per-lane column, 8-element runs at kBase+8*hi and kBase+16+8*hi.
static __device__ __forceinline__ v16bf load_b16(const bf16_t* __restrict__ colPtr,
                                                 int kBase, int hi) {
  v8bf c0 = *(const v8bf*)(colPtr + kBase + 8 * hi);
  v8bf c1 = *(const v8bf*)(colPtr + kBase + 16 + 8 * hi);
  return __builtin_shufflevector(c0, c1, 0, 1, 2, 3, 4, 5, 6, 7,
                                 8, 9, 10, 11, 12, 13, 14, 15);
}

// gfx1250 async DMA: copy 8 bytes global -> LDS, tracked by ASYNCcnt.
static __device__ __forceinline__ void async_copy8(bf16_t* lds, const bf16_t* g) {
  unsigned ldsOff = (unsigned)(unsigned long long)(uintptr_t)lds;
  unsigned long long ga = (unsigned long long)(uintptr_t)g;
  asm volatile("global_load_async_to_lds_b64 %0, %1, off"
               :: "v"(ldsOff), "v"(ga) : "memory");
}
static __device__ __forceinline__ void wait_async0() {
  asm volatile("s_wait_asynccnt 0x0" ::: "memory");
}

static __device__ __forceinline__ float sigmoidf_(float x) {
  return 1.0f / (1.0f + __expf(-x));
}

// ---------------------------------------------------------------------------
// One K-phase of the step GEMM. Weight slab (A) is DMA'd global->LDS into a
// double buffer (one barrier per K-step); B column is private per wave and
// pipelined one K-step ahead in registers.
// ---------------------------------------------------------------------------
static __device__ __forceinline__ void gemm_phase(
    const bf16_t* __restrict__ W0, const bf16_t* __restrict__ W1,
    const bf16_t* __restrict__ W2, const bf16_t* __restrict__ W3,
    int ldk, int K, const bf16_t* __restrict__ colPtr,
    bf16_t* Ash, int rowBase, int sg, int sr, int sc, int n, int hi,
    v8f& accg, v8f& acci, v8f& accf, v8f& acco) {
  const bf16_t* Wsel = (sg == 0) ? W0 : (sg == 1) ? W1 : (sg == 2) ? W2 : W3;
  const bf16_t* gsrc = Wsel + (size_t)(rowBase + sr) * ldk + sc * 4;
  const int sOff = (sg * 16 + sr) * AROW + sc * 4;   // chunk offset within slab

  // prologue: DMA slab for k=0 into buffer 0; prefetch B column for k=0
  async_copy8(Ash + sOff, gsrc);
  v16bf bt = load_b16(colPtr, 0, hi);
  wait_async0();
  __syncthreads();

  int cur = 0;
  for (int k = 0; k < K; k += 32) {
    if (k + 32 < K) {                                // DMA next slab -> buf^1
      async_copy8(Ash + (cur ^ 1) * SLAB + sOff, gsrc + k + 32);
      __builtin_prefetch(colPtr + k + 256, 0, 1);
    }
    v16bf btn = bt;
    if (k + 32 < K) btn = load_b16(colPtr, k + 32, hi);

    const bf16_t* base = Ash + cur * SLAB;
    accg = wmma_bf16(load_a16(base + (0 * 16 + n) * AROW, hi), bt, accg);
    acci = wmma_bf16(load_a16(base + (1 * 16 + n) * AROW, hi), bt, acci);
    accf = wmma_bf16(load_a16(base + (2 * 16 + n) * AROW, hi), bt, accf);
    acco = wmma_bf16(load_a16(base + (3 * 16 + n) * AROW, hi), bt, acco);

    wait_async0();                                   // this wave's DMA landed
    __syncthreads();                                 // everyone's DMA landed
    cur ^= 1;
    bt = btn;
  }
}

// ---------------------------------------------------------------------------
// f32 -> bf16 conversion
// ---------------------------------------------------------------------------
__global__ void cvt_f32_bf16(const float* __restrict__ src,
                             bf16_t* __restrict__ dst, int n) {
  int i = blockIdx.x * blockDim.x + threadIdx.x;
  if (i < n) dst[i] = (bf16_t)src[i];
}

// zero h (bf16, [B][H]) and c (f32, [B][H])
__global__ void init_state(bf16_t* __restrict__ h, float* __restrict__ c, int n) {
  int i = blockIdx.x * blockDim.x + threadIdx.x;
  if (i < n) {
    h[i] = (bf16_t)0.0f;
    c[i] = 0.0f;
  }
}

// ---------------------------------------------------------------------------
// One LSTM timestep.
// grid: 64 blocks (one per 16-row H tile); block: (32,16) = 16 waves.
// Wave ty computes the 16(H) x 16(B) tile for batch cols [16*ty, 16*ty+16),
// fusing the input projection (embed gather, K over D) with the recurrent
// GEMM (K over H). Weight slabs DMA'd once per K-step into LDS.
// ---------------------------------------------------------------------------
__global__ __launch_bounds__(512) void lstm_step(
    const int* __restrict__ x, const bf16_t* __restrict__ embB,
    const bf16_t* __restrict__ Wgx, const bf16_t* __restrict__ Wix,
    const bf16_t* __restrict__ Wfx, const bf16_t* __restrict__ Wox,
    const bf16_t* __restrict__ Wgh, const bf16_t* __restrict__ Wih,
    const bf16_t* __restrict__ Wfh, const bf16_t* __restrict__ Woh,
    const float* __restrict__ bg, const float* __restrict__ bi,
    const float* __restrict__ bf_, const float* __restrict__ bo,
    const bf16_t* __restrict__ hPrev, bf16_t* __restrict__ hNext,
    float* __restrict__ c, int s) {
  __shared__ bf16_t Ash[2 * SLAB];           // double-buffered weight slab (10 KB)

  const int lane = threadIdx.x;              // 0..31
  const int ty = threadIdx.y;                // 0..15  (col tile)
  const int tid = ty * 32 + lane;            // 0..511
  const int rowBase = blockIdx.x * 16;       // H tile
  const int n = lane & 15;
  const int hi = lane >> 4;
  const int b = ty * 16 + n;                 // batch column for this lane

  // cooperative DMA ids: 512 threads x 8B chunk = 4*16*32 bf16 slab
  const int sg = tid >> 7;                   // gate 0..3
  const int sr = (tid >> 3) & 15;            // row within tile
  const int sc = tid & 7;                    // 4-element chunk within 32-wide K

  v8f accg = {}, acci = {}, accf = {}, acco = {};

  // ---- input projection: K over D, B column = embed[x[b,s], :]
  {
    const int idx = x[b * Sn + s];
    const bf16_t* xcol = embB + (size_t)idx * Dn;
    gemm_phase(Wgx, Wix, Wfx, Wox, Dn, Dn, xcol, Ash, rowBase,
               sg, sr, sc, n, hi, accg, acci, accf, acco);
  }
  // ---- recurrent projection: K over H, B column = hPrev[b, :]
  {
    const bf16_t* hcol = hPrev + (size_t)b * Hn;
    gemm_phase(Wgh, Wih, Wfh, Woh, Hn, Hn, hcol, Ash, rowBase,
               sg, sr, sc, n, hi, accg, acci, accf, acco);
  }

  // ---- elementwise gate math; C/D layout: lane col = b, VGPR v -> row r0+v
  const int r0 = rowBase + 8 * hi;
  v8f bgv = *(const v8f*)(bg + r0);
  v8f biv = *(const v8f*)(bi + r0);
  v8f bfv = *(const v8f*)(bf_ + r0);
  v8f bov = *(const v8f*)(bo + r0);
  v8f cv = *(const v8f*)(c + (size_t)b * Hn + r0);
  v8bf hb;
#pragma unroll
  for (int v = 0; v < 8; ++v) {
    float g = tanhf(accg[v] + bgv[v]);
    float ii = sigmoidf_(acci[v] + biv[v]);
    float ff = sigmoidf_(accf[v] + bfv[v]);
    float oo = sigmoidf_(acco[v] + bov[v]);
    float cn = g * ii + cv[v] * ff;
    cv[v] = cn;
    hb[v] = (bf16_t)(tanhf(cn) * oo);
  }
  *(v8f*)(c + (size_t)b * Hn + r0) = cv;       // in-place: tile-private
  *(v8bf*)(hNext + (size_t)b * Hn + r0) = hb;  // ping-pong buffer
}

// ---------------------------------------------------------------------------
// Final projection: logits[b][cls] = W_ph @ h + b_p   (WMMA, f32 out)
// grid: 32 blocks, block (32,8) => 256 waves = (C/16)*(B/16) tiles
// ---------------------------------------------------------------------------
__global__ __launch_bounds__(256) void proj_kernel(
    const bf16_t* __restrict__ Wph, const bf16_t* __restrict__ h,
    const float* __restrict__ bp, float* __restrict__ logits) {
  const int lane = threadIdx.x;
  const int w = blockIdx.x * 8 + threadIdx.y;  // 0..255
  const int rowTile = w & 15;                  // C/16
  const int colTile = w >> 4;                  // B/16
  const int n = lane & 15;
  const int hi = lane >> 4;
  const int m = rowTile * 16 + n;              // class row
  const int b = colTile * 16 + n;              // batch col

  v8f acc = {};
  const bf16_t* a = Wph + (size_t)m * Hn;
  const bf16_t* hcol = h + (size_t)b * Hn;
#pragma unroll 4
  for (int k = 0; k < Hn; k += 32) {
    acc = wmma_bf16(load_a16(a + k, hi), load_b16(hcol, k, hi), acc);
  }
  const int r0 = rowTile * 16 + 8 * hi;
  v8f bpv = *(const v8f*)(bp + r0);
  v8f outv;
#pragma unroll
  for (int v = 0; v < 8; ++v) outv[v] = acc[v] + bpv[v];
  *(v8f*)(logits + (size_t)b * Cn + r0) = outv;
}

// ---------------------------------------------------------------------------
// log_softmax over class dim; one block per batch row. out = [B][C] (== y.T)
// ---------------------------------------------------------------------------
__global__ __launch_bounds__(256) void logsoftmax_kernel(
    const float* __restrict__ logits, float* __restrict__ out) {
  __shared__ float sm[Cn];
  const int b = blockIdx.x;
  const int t = threadIdx.x;
  float v = logits[(size_t)b * Cn + t];
  sm[t] = v;
  __syncthreads();
  for (int o = Cn / 2; o > 0; o >>= 1) {
    if (t < o) sm[t] = fmaxf(sm[t], sm[t + o]);
    __syncthreads();
  }
  float mx = sm[0];
  __syncthreads();
  sm[t] = __expf(v - mx);
  __syncthreads();
  for (int o = Cn / 2; o > 0; o >>= 1) {
    if (t < o) sm[t] = sm[t] + sm[t + o];
    __syncthreads();
  }
  float lse = __logf(sm[0]);
  out[(size_t)b * Cn + t] = v - mx - lse;
}

// ---------------------------------------------------------------------------
extern "C" void kernel_launch(void* const* d_in, const int* in_sizes, int n_in,
                              void* d_out, int out_size, void* d_ws, size_t ws_size,
                              hipStream_t stream) {
  const int*   x     = (const int*)d_in[0];
  const float* embed = (const float*)d_in[1];
  const float* Wgx_f = (const float*)d_in[2];
  const float* Wgh_f = (const float*)d_in[3];
  const float* Wix_f = (const float*)d_in[4];
  const float* Wih_f = (const float*)d_in[5];
  const float* Wfx_f = (const float*)d_in[6];
  const float* Wfh_f = (const float*)d_in[7];
  const float* Wox_f = (const float*)d_in[8];
  const float* Woh_f = (const float*)d_in[9];
  const float* Wph_f = (const float*)d_in[10];
  const float* bg    = (const float*)d_in[11];
  const float* bi    = (const float*)d_in[12];
  const float* bf_   = (const float*)d_in[13];
  const float* bo    = (const float*)d_in[14];
  const float* bp    = (const float*)d_in[15];
  float* out = (float*)d_out;

  // ---- workspace partitioning (~13.5 MB)
  char* ws = (char*)d_ws;
  size_t off = 0;
  auto alloc_bf = [&](size_t nelem) {
    bf16_t* p = (bf16_t*)(ws + off);
    off += nelem * sizeof(bf16_t);
    return p;
  };
  auto alloc_f = [&](size_t nelem) {
    float* p = (float*)(ws + off);
    off += nelem * sizeof(float);
    return p;
  };
  bf16_t* embB = alloc_bf((size_t)Dn * Dn);
  bf16_t* WgxB = alloc_bf((size_t)Hn * Dn);
  bf16_t* WixB = alloc_bf((size_t)Hn * Dn);
  bf16_t* WfxB = alloc_bf((size_t)Hn * Dn);
  bf16_t* WoxB = alloc_bf((size_t)Hn * Dn);
  bf16_t* WghB = alloc_bf((size_t)Hn * Hn);
  bf16_t* WihB = alloc_bf((size_t)Hn * Hn);
  bf16_t* WfhB = alloc_bf((size_t)Hn * Hn);
  bf16_t* WohB = alloc_bf((size_t)Hn * Hn);
  bf16_t* WphB = alloc_bf((size_t)Cn * Hn);
  bf16_t* hA   = alloc_bf((size_t)Bn * Hn);
  bf16_t* hB   = alloc_bf((size_t)Bn * Hn);
  float*  cSt  = alloc_f((size_t)Bn * Hn);
  float*  logits = alloc_f((size_t)Bn * Cn);

  auto cvt = [&](const float* src, bf16_t* dst, int n) {
    cvt_f32_bf16<<<(n + 255) / 256, 256, 0, stream>>>(src, dst, n);
  };
  cvt(embed, embB, Dn * Dn);
  cvt(Wgx_f, WgxB, Hn * Dn);
  cvt(Wix_f, WixB, Hn * Dn);
  cvt(Wfx_f, WfxB, Hn * Dn);
  cvt(Wox_f, WoxB, Hn * Dn);
  cvt(Wgh_f, WghB, Hn * Hn);
  cvt(Wih_f, WihB, Hn * Hn);
  cvt(Wfh_f, WfhB, Hn * Hn);
  cvt(Woh_f, WohB, Hn * Hn);
  cvt(Wph_f, WphB, Cn * Hn);

  init_state<<<(Bn * Hn + 255) / 256, 256, 0, stream>>>(hA, cSt, Bn * Hn);

  bf16_t* hPrev = hA;
  bf16_t* hNext = hB;
  for (int s = 0; s < Sn; ++s) {
    lstm_step<<<dim3(64), dim3(32, 16), 0, stream>>>(
        x, embB, WgxB, WixB, WfxB, WoxB, WghB, WihB, WfhB, WohB,
        bg, bi, bf_, bo, hPrev, hNext, cSt, s);
    bf16_t* t = hPrev; hPrev = hNext; hNext = t;
  }

  proj_kernel<<<dim3(32), dim3(32, 8), 0, stream>>>(WphB, hPrev, bp, logits);
  logsoftmax_kernel<<<Bn, Cn, 0, stream>>>(logits, out);
}
